// TransformerLayer_13915694039431
// MI455X (gfx1250) — matrix operand order repaired and checked by reading, not compiled, not run on hardware
//
#include <hip/hip_runtime.h>
#include <hip/hip_bf16.h>
#include <math.h>

// Problem constants (match reference)
#define B_   4
#define S_   2048
#define HID_ 1024
#define H_   4
#define D_   256
#define EPS_ 1e-5f

typedef __bf16 bf16_t;
typedef __attribute__((ext_vector_type(16))) __bf16 v16bf;
typedef __attribute__((ext_vector_type(8)))  float  v8f;

union Frag16 { v16bf v; unsigned int u[8]; bf16_t h[16]; };
union U4     { uint4 q; unsigned int u[4]; bf16_t h[8]; };

// ---------------------------------------------------------------------------
// CDNA5 async global->LDS staging (GLOBAL_LOAD_ASYNC_TO_LDS_B128, ASYNCcnt).
// Probe-confirmed signature: (int4 AS(1)*, int4 AS(3)*, imm offset, imm cpol).
// Guarded so the file still compiles if a toolchain lacks the builtins.
// ---------------------------------------------------------------------------
#if __has_builtin(__builtin_amdgcn_global_load_async_to_lds_b128) && \
    __has_builtin(__builtin_amdgcn_s_wait_asynccnt)
#define HAVE_ASYNC_LDS 1
#else
#define HAVE_ASYNC_LDS 0
#endif

typedef int v4i_vs __attribute__((vector_size(16)));
typedef __attribute__((address_space(1))) v4i_vs gbl_v4i;
typedef __attribute__((address_space(3))) v4i_vs lds_v4i;

static __device__ __forceinline__ void stage_b128(const bf16_t* g, bf16_t* l) {
#if HAVE_ASYNC_LDS
  __builtin_amdgcn_global_load_async_to_lds_b128((gbl_v4i*)g, (lds_v4i*)l, 0, 0);
#else
  U4 t;
  t.q = *reinterpret_cast<const uint4*>(g);
  unsigned int* dst = reinterpret_cast<unsigned int*>(l);
  dst[0] = t.u[0]; dst[1] = t.u[1]; dst[2] = t.u[2]; dst[3] = t.u[3];
#endif
}

static __device__ __forceinline__ void stage_wait() {
#if HAVE_ASYNC_LDS
  __builtin_amdgcn_s_wait_asynccnt(0);
#endif
}

// ---------------------------------------------------------------------------
// WMMA helpers.  Fragment layouts per CDNA5 ISA 7.12.2 (wave32):
//   A (16x32 bf16): lane L -> M = L&15; half=L>>4; VGPR j pairs:
//        j<4:  K = 8*half + 2j (+1) ;  j>=4: K = 16 + 8*half + 2(j-4) (+1)
//   B (32x16 bf16): lane L -> N = L&15; same K pairing as A
//   C/D (16x16 f32): element r -> M = r + 8*half, N = lane&15
// ---------------------------------------------------------------------------
static __device__ __forceinline__ v16bf lds_afrag(const bf16_t* base, int stride) {
  const int lane = threadIdx.x & 31;
  const int m = lane & 15, half = lane >> 4;
  Frag16 f;
  const bf16_t* row = base + m * stride;
#pragma unroll
  for (int j = 0; j < 8; ++j) {
    const int k = (j < 4) ? (8 * half + 2 * j) : (16 + 8 * half + 2 * (j - 4));
    f.u[j] = *reinterpret_cast<const unsigned int*>(row + k);
  }
  return f.v;
}

// base is laid out [n][k] with k contiguous (i.e. B^T), stride in elements
static __device__ __forceinline__ v16bf lds_bfrag(const bf16_t* base, int stride) {
  const int lane = threadIdx.x & 31;
  const int n = lane & 15, half = lane >> 4;
  Frag16 f;
  const bf16_t* col = base + n * stride;
#pragma unroll
  for (int j = 0; j < 8; ++j) {
    const int k = (j < 4) ? (8 * half + 2 * j) : (16 + 8 * half + 2 * (j - 4));
    f.u[j] = *reinterpret_cast<const unsigned int*>(col + k);
  }
  return f.v;
}

static __device__ __forceinline__ v8f wmma_bf16(v16bf a, v16bf b, v8f c) {
  return __builtin_amdgcn_wmma_f32_16x16x32_bf16(false, a, false, b, (short)0, c,
                                                 false, false);
}

// ---------------------------------------------------------------------------
// f32 -> bf16 conversion (grid-stride)
// ---------------------------------------------------------------------------
__global__ void cvt_f32_bf16(const float* __restrict__ in, bf16_t* __restrict__ out,
                             int n) {
  int i = blockIdx.x * blockDim.x + threadIdx.x;
  const int stride = gridDim.x * blockDim.x;
  for (; i < n; i += stride) out[i] = (bf16_t)in[i];
}

// ---------------------------------------------------------------------------
// Tiled WMMA GEMM:  C[M,N] = A[M,K] (bf16, row major) @ W[K,N] (bf16, row major) + bias
// MODE 0: scatter bf16 to [B,H,S,D] with scale (QKV)
// MODE 1: exact GELU -> bf16 row-major
// MODE 2: f32 row-major
// Block: 128 threads (4 waves), 64x64 tile, 32 K-step, 2x2 16x16 frags/wave.
// A tile staged with async global->LDS (stride 40: 16B-aligned, conflict-free);
// B tile staged synchronously because it is transposed in flight.
// ---------------------------------------------------------------------------
#define SA_STR 40
template <int MODE>
__global__ __launch_bounds__(128) void gemm_wmma(
    const bf16_t* __restrict__ A, const bf16_t* __restrict__ W,
    const float* __restrict__ bias, float scale, bf16_t* __restrict__ outb,
    float* __restrict__ outf, int M, int K, int N) {
  __shared__ bf16_t sA[64 * SA_STR];  // [m][k]
  __shared__ bf16_t sB[64 * 34];      // [n][k] (transposed), padded stride 34

  const int tid = threadIdx.x;
  const int wave = tid >> 5, lane = tid & 31;
  const int wm = wave >> 1, wn = wave & 1;
  const int tn = blockIdx.x * 64, tm = blockIdx.y * 64;

  const v8f vzero = {0.f, 0.f, 0.f, 0.f, 0.f, 0.f, 0.f, 0.f};
  v8f acc[2][2] = {{vzero, vzero}, {vzero, vzero}};

  const int rowA = tid >> 1, colA = (tid & 1) * 16;  // 64 rows x 32 cols
  const int rowB = tid >> 2, colB = (tid & 3) * 16;  // 32 rows x 64 cols

  for (int kk = 0; kk < K; kk += 32) {
    // --- stage A (64x32) via async DMA to LDS ---
    {
      const bf16_t* gp = A + (size_t)(tm + rowA) * K + kk + colA;
      bf16_t* lp = sA + rowA * SA_STR + colA;
      stage_b128(gp, lp);
      stage_b128(gp + 8, lp + 8);
    }
    // --- stage B (32x64) transposed into [n][k] ---
    {
      const bf16_t* gp = W + (size_t)(kk + rowB) * N + tn + colB;
      U4 b0, b1;
      b0.q = reinterpret_cast<const uint4*>(gp)[0];
      b1.q = reinterpret_cast<const uint4*>(gp)[1];
#pragma unroll
      for (int e = 0; e < 8; ++e) sB[(colB + e) * 34 + rowB] = b0.h[e];
#pragma unroll
      for (int e = 0; e < 8; ++e) sB[(colB + 8 + e) * 34 + rowB] = b1.h[e];
    }
    if (kk + 32 < K) {  // prefetch next tiles (global_prefetch_b8)
      __builtin_prefetch(A + (size_t)(tm + rowA) * K + kk + 32 + colA, 0, 1);
      __builtin_prefetch(W + (size_t)(kk + 32 + rowB) * N + tn + colB, 0, 1);
    }
    stage_wait();
    __syncthreads();

    v16bf afr[2], bfr[2];
#pragma unroll
    for (int mi = 0; mi < 2; ++mi)
      afr[mi] = lds_afrag(sA + (wm * 32 + mi * 16) * SA_STR, SA_STR);
#pragma unroll
    for (int ni = 0; ni < 2; ++ni)
      bfr[ni] = lds_bfrag(sB + (wn * 32 + ni * 16) * 34, 34);
#pragma unroll
    for (int mi = 0; mi < 2; ++mi)
#pragma unroll
      for (int ni = 0; ni < 2; ++ni)
        acc[mi][ni] = wmma_bf16(afr[mi], bfr[ni], acc[mi][ni]);
    __syncthreads();
  }

  const int half = lane >> 4, ln = lane & 15;
#pragma unroll
  for (int mi = 0; mi < 2; ++mi)
#pragma unroll
    for (int ni = 0; ni < 2; ++ni) {
      const int ncol = tn + wn * 32 + ni * 16 + ln;
      const float bv = bias[ncol];
#pragma unroll
      for (int r = 0; r < 8; ++r) {
        const int mrow = tm + wm * 32 + mi * 16 + r + 8 * half;
        float v = acc[mi][ni][r] + bv;
        if (MODE == 0) {
          const int b = mrow >> 11, s = mrow & (S_ - 1);
          const int h = ncol >> 8, d = ncol & (D_ - 1);
          outb[(((size_t)(b * H_ + h)) * S_ + s) * D_ + d] = (bf16_t)(v * scale);
        } else if (MODE == 1) {
          const float g = 0.5f * v * (1.f + erff(v * 0.70710678118654752f));
          outb[(size_t)mrow * N + ncol] = (bf16_t)g;
        } else {
          outf[(size_t)mrow * N + ncol] = v;
        }
      }
    }
}

// ---------------------------------------------------------------------------
// Causal flash attention, WMMA.  Block = 64 threads (2 waves), 32 query rows
// per block (16 per wave).  Keys processed in 32-wide chunks staged in LDS.
// q/k tiles staged with async global->LDS (stride 264: 16B-aligned rows,
// conflict-free fragment reads); v staged synchronously (transposed in flight).
// q is pre-scaled by 1/sqrt(HID).  att_max = 1/l (online softmax identity).
// ---------------------------------------------------------------------------
#define QK_STR 264
__global__ __launch_bounds__(64) void attn_kernel(
    const bf16_t* __restrict__ q, const bf16_t* __restrict__ k,
    const bf16_t* __restrict__ v, const int* __restrict__ lengths,
    float* __restrict__ attn_out, float* __restrict__ att_max) {
  __shared__ bf16_t qs[32 * QK_STR];  // [qrow][d]
  __shared__ bf16_t ks[32 * QK_STR];  // [key][d]   (== B^T for scores)
  __shared__ bf16_t vts[256 * 34];    // [d][key]   (B^T for P@V)
  __shared__ bf16_t as[2][16 * 34];   // per-wave P tile [qrow][key]

  const int tid = threadIdx.x, wave = tid >> 5, lane = tid & 31;
  const int half = lane >> 4, ln = lane & 15;
  const int q0 = blockIdx.x * 32;
  const int bh = blockIdx.y;  // b*H + h
  const int q0w = q0 + wave * 16;

  // --- stage q tile (32 x 256) async ---
  {
    const int rowQ = tid >> 1, col0 = (tid & 1) * 128;
    const bf16_t* gp = q + ((size_t)bh * S_ + q0 + rowQ) * D_ + col0;
    bf16_t* lp = qs + rowQ * QK_STR + col0;
#pragma unroll
    for (int e = 0; e < 16; ++e) stage_b128(gp + e * 8, lp + e * 8);
  }

  const v8f vzero = {0.f, 0.f, 0.f, 0.f, 0.f, 0.f, 0.f, 0.f};
  v8f acc[16];
#pragma unroll
  for (int i = 0; i < 16; ++i) acc[i] = vzero;
  float m[8], l[8];
#pragma unroll
  for (int r = 0; r < 8; ++r) { m[r] = -INFINITY; l[r] = 0.f; }

  const int nch = (q0 + 32) >> 5;  // causal: only keys <= q0+31
  for (int kc = 0; kc < nch; ++kc) {
    __syncthreads();
    // --- stage k chunk (32 x 256, async) and v chunk transposed (sync) ---
    {
      const int rowK = tid >> 1, col0 = (tid & 1) * 128;
      const bf16_t* gk = k + ((size_t)bh * S_ + kc * 32 + rowK) * D_ + col0;
      const bf16_t* gv = v + ((size_t)bh * S_ + kc * 32 + rowK) * D_ + col0;
      bf16_t* lk = ks + rowK * QK_STR + col0;
#pragma unroll
      for (int e = 0; e < 16; ++e) stage_b128(gk + e * 8, lk + e * 8);
#pragma unroll
      for (int e = 0; e < 16; ++e) {
        U4 t; t.q = reinterpret_cast<const uint4*>(gv)[e];
#pragma unroll
        for (int x = 0; x < 8; ++x)
          vts[(col0 + e * 8 + x) * 34 + rowK] = t.h[x];
      }
    }
    stage_wait();
    __syncthreads();

    // --- scores: S = q @ k^T  (16x32 per wave, K=256) ---
    v8f sc0 = vzero, sc1 = vzero;
#pragma unroll
    for (int dk = 0; dk < 8; ++dk) {
      const v16bf qa = lds_afrag(qs + (wave * 16) * QK_STR + dk * 32, QK_STR);
      const v16bf kf0 = lds_bfrag(ks + 0 * QK_STR + dk * 32, QK_STR);
      const v16bf kf1 = lds_bfrag(ks + 16 * QK_STR + dk * 32, QK_STR);
      sc0 = wmma_bf16(qa, kf0, sc0);
      sc1 = wmma_bf16(qa, kf1, sc1);
    }

    // --- online softmax (per query row r + 8*half) ---
    bf16_t* aw = &as[wave][0];
    float scl[8];
#pragma unroll
    for (int r = 0; r < 8; ++r) {
      const int qrow = q0w + r + 8 * half;
      const int key0 = kc * 32 + ln;
      float s0 = (key0 <= qrow) ? sc0[r] : -INFINITY;
      float s1 = (key0 + 16 <= qrow) ? sc1[r] : -INFINITY;
      float mx = fmaxf(s0, s1);
      mx = fmaxf(mx, __shfl_xor(mx, 1));
      mx = fmaxf(mx, __shfl_xor(mx, 2));
      mx = fmaxf(mx, __shfl_xor(mx, 4));
      mx = fmaxf(mx, __shfl_xor(mx, 8));
      const float mn = fmaxf(m[r], mx);
      float p0, p1, scale;
      if (mn == -INFINITY) { p0 = 0.f; p1 = 0.f; scale = 1.f; }
      else { p0 = __expf(s0 - mn); p1 = __expf(s1 - mn); scale = __expf(m[r] - mn); }
      float ps = p0 + p1;
      ps += __shfl_xor(ps, 1);
      ps += __shfl_xor(ps, 2);
      ps += __shfl_xor(ps, 4);
      ps += __shfl_xor(ps, 8);
      l[r] = l[r] * scale + ps;
      m[r] = mn;
      scl[r] = scale;
      aw[(r + 8 * half) * 34 + ln] = (bf16_t)p0;
      aw[(r + 8 * half) * 34 + ln + 16] = (bf16_t)p1;
    }

    // --- O = O*scale + P @ V  (K = 32 keys, N = 256 d) ---
    const v16bf afr = lds_afrag(aw, 34);
#pragma unroll
    for (int nb = 0; nb < 16; ++nb) {
      const v16bf vf = lds_bfrag(vts + (nb * 16) * 34, 34);
#pragma unroll
      for (int r = 0; r < 8; ++r) acc[nb][r] *= scl[r];
      acc[nb] = wmma_bf16(afr, vf, acc[nb]);
    }
  }

  // --- finalize: divide by l, apply row-length mask, emit att_max = 1/l ---
  const int b = bh >> 2;  // H_ == 4
  const int len = lengths[b];
#pragma unroll
  for (int r = 0; r < 8; ++r) {
    const int qrow = q0w + r + 8 * half;
    const float inv = (l[r] > 0.f) ? 1.f / l[r] : 0.f;
    const bool valid = qrow < len;
    const float outscale = valid ? inv : 0.f;
    if (ln == 0) att_max[(size_t)bh * S_ + qrow] = outscale;
#pragma unroll
    for (int nb = 0; nb < 16; ++nb)
      attn_out[((size_t)bh * S_ + qrow) * D_ + nb * 16 + ln] =
          acc[nb][r] * outscale;
  }
}

// ---------------------------------------------------------------------------
// Fused residual + LayerNorm over rows of 1024.  Block = 128 threads.
// ---------------------------------------------------------------------------
template <bool STORE_BF16>
__global__ __launch_bounds__(128) void residual_ln(
    const float* __restrict__ a, const float* __restrict__ b,
    const float* __restrict__ g, const float* __restrict__ be,
    float* __restrict__ outf, bf16_t* __restrict__ outb) {
  __shared__ float red[8];
  const int row = blockIdx.x, tid = threadIdx.x;
  const int wave = tid >> 5, lane = tid & 31;
  const size_t base = (size_t)row * HID_;
  float x[8];
  float s = 0.f, sq = 0.f;
#pragma unroll
  for (int e = 0; e < 8; ++e) {
    const int c = tid * 8 + e;
    const float val = a[base + c] + b[base + c];
    x[e] = val; s += val; sq += val * val;
  }
#pragma unroll
  for (int off = 1; off < 32; off <<= 1) {
    s += __shfl_xor(s, off);
    sq += __shfl_xor(sq, off);
  }
  if (lane == 0) { red[wave] = s; red[4 + wave] = sq; }
  __syncthreads();
  s = red[0] + red[1] + red[2] + red[3];
  sq = red[4] + red[5] + red[6] + red[7];
  const float mean = s * (1.f / HID_);
  const float var = sq * (1.f / HID_) - mean * mean;
  const float rstd = rsqrtf(var + EPS_);
#pragma unroll
  for (int e = 0; e < 8; ++e) {
    const int c = tid * 8 + e;
    const float y = (x[e] - mean) * rstd * g[c] + be[c];
    outf[base + c] = y;
    if (STORE_BF16) outb[base + c] = (bf16_t)y;
  }
}

__global__ void copy_lengths(const int* __restrict__ lengths, int* __restrict__ out) {
  if (threadIdx.x < B_) out[threadIdx.x] = lengths[threadIdx.x];
}

// ---------------------------------------------------------------------------
extern "C" void kernel_launch(void* const* d_in, const int* in_sizes, int n_in,
                              void* d_out, int out_size, void* d_ws, size_t ws_size,
                              hipStream_t stream) {
  const float* x       = (const float*)d_in[0];
  const int*   lengths = (const int*)d_in[1];
  const float* Wq = (const float*)d_in[3];
  const float* bq = (const float*)d_in[4];
  const float* Wk = (const float*)d_in[5];
  const float* bk = (const float*)d_in[6];
  const float* Wv = (const float*)d_in[7];
  const float* bv = (const float*)d_in[8];
  const float* W1 = (const float*)d_in[9];
  const float* b1 = (const float*)d_in[10];
  const float* W2 = (const float*)d_in[11];
  const float* b2 = (const float*)d_in[12];
  const float* g1 = (const float*)d_in[13];
  const float* be1 = (const float*)d_in[14];
  const float* g2 = (const float*)d_in[15];
  const float* be2 = (const float*)d_in[16];

  const size_t NX = (size_t)B_ * S_ * HID_;  // 8,388,608
  const size_t NW = (size_t)HID_ * HID_;     // 1,048,576

  char* ws = (char*)d_ws;
  size_t off = 0;
  auto alloc = [&](size_t bytes) -> char* {
    char* p = ws + off;
    off += (bytes + 255) & ~(size_t)255;
    return p;
  };
  bf16_t* xb   = (bf16_t*)alloc(NX * 2);
  bf16_t* wqb  = (bf16_t*)alloc(NW * 2);
  bf16_t* wkb  = (bf16_t*)alloc(NW * 2);
  bf16_t* wvb  = (bf16_t*)alloc(NW * 2);
  bf16_t* w1b  = (bf16_t*)alloc(NW * 2);
  bf16_t* w2b  = (bf16_t*)alloc(NW * 2);
  bf16_t* qb   = (bf16_t*)alloc(NX * 2);
  bf16_t* kb   = (bf16_t*)alloc(NX * 2);
  bf16_t* vb   = (bf16_t*)alloc(NX * 2);
  float*  attn = (float*)alloc(NX * 4);
  float*  o1n  = (float*)alloc(NX * 4);
  bf16_t* o1nb = (bf16_t*)alloc(NX * 2);
  bf16_t* ff1  = (bf16_t*)alloc(NX * 2);
  float*  ff2  = (float*)alloc(NX * 4);

  float* yout   = (float*)d_out;               // [B,S,HID]
  int*   lenout = (int*)((float*)d_out + NX);  // [B]
  float* amout  = (float*)d_out + NX + B_;     // [B,H,S]

  // 1. bf16 conversions
  cvt_f32_bf16<<<4096, 256, 0, stream>>>(x, xb, (int)NX);
  cvt_f32_bf16<<<1024, 256, 0, stream>>>(Wq, wqb, (int)NW);
  cvt_f32_bf16<<<1024, 256, 0, stream>>>(Wk, wkb, (int)NW);
  cvt_f32_bf16<<<1024, 256, 0, stream>>>(Wv, wvb, (int)NW);
  cvt_f32_bf16<<<1024, 256, 0, stream>>>(W1, w1b, (int)NW);
  cvt_f32_bf16<<<1024, 256, 0, stream>>>(W2, w2b, (int)NW);

  const int M = B_ * S_, K = HID_, N = HID_;
  dim3 gemm_grid(N / 64, M / 64), gemm_block(128);

  // 2. QKV projections (q pre-scaled by 1/sqrt(HID) = 1/32)
  gemm_wmma<0><<<gemm_grid, gemm_block, 0, stream>>>(xb, wqb, bq, 0.03125f, qb,
                                                     nullptr, M, K, N);
  gemm_wmma<0><<<gemm_grid, gemm_block, 0, stream>>>(xb, wkb, bk, 1.0f, kb,
                                                     nullptr, M, K, N);
  gemm_wmma<0><<<gemm_grid, gemm_block, 0, stream>>>(xb, wvb, bv, 1.0f, vb,
                                                     nullptr, M, K, N);

  // 3. causal attention + att_max
  attn_kernel<<<dim3(S_ / 32, B_ * H_), 64, 0, stream>>>(qb, kb, vb, lengths,
                                                         attn, amout);

  // 4. out_1 = x + attn (flat == the reference's reshape bug); LN1
  residual_ln<true><<<M, 128, 0, stream>>>(x, attn, g1, be1, o1n, o1nb);

  // 5. FFN
  gemm_wmma<1><<<gemm_grid, gemm_block, 0, stream>>>(o1nb, w1b, b1, 1.0f, ff1,
                                                     nullptr, M, K, N);
  gemm_wmma<2><<<gemm_grid, gemm_block, 0, stream>>>(ff1, w2b, b2, 1.0f, nullptr,
                                                     ff2, M, K, N);

  // 6. y = LN2(out_1n + ff) -> d_out
  residual_ln<false><<<M, 128, 0, stream>>>(o1n, ff2, g2, be2, yout, nullptr);

  // 7. lengths passthrough
  copy_lengths<<<1, 32, 0, stream>>>(lengths, lenout);
}
